// NegativeSamplingBCELoss_7687991459998
// MI455X (gfx1250) — matrix operand order repaired and checked by reading, not compiled, not run on hardware
//
#include <hip/hip_runtime.h>
#include <cmath>

// ---------------------------------------------------------------------------
// NegativeSamplingBCELoss for MI455X (gfx1250, wave32, WMMA)
//   B=8192 rows, C=2048 classes (derived from in_sizes at launch).
//   Heavy op: posSim = pos @ similarity -> f16 WMMA GEMM, f32 accumulate.
//   Pipeline: [transpose sim -> f16 simT] -> [WMMA GEMM, double-buffered LDS]
//             -> [per-row Gumbel-top-k + masked BCE] -> [finalize scalar]
// ---------------------------------------------------------------------------

typedef _Float16 v16h __attribute__((ext_vector_type(16)));
typedef _Float16 v8h  __attribute__((ext_vector_type(8)));
typedef float    v8f  __attribute__((ext_vector_type(8)));

#define TM   128   // workgroup M tile
#define TN   128   // workgroup N tile
#define TKS  32    // K step (one f16 WMMA K)
#define LPAD 8     // LDS pad (row stride 40 halfs = 80 B, 16 B multiple)

// ---------------------------------------------------------------------------
// Kernel 0: simT[n][k] = (f16) sim[k][n]   (tiled transpose + convert)
// ---------------------------------------------------------------------------
__global__ __launch_bounds__(256) void transpose_f16_kernel(
    const float* __restrict__ sim, _Float16* __restrict__ simT, int C)
{
    __shared__ float tile[32][33];
    const int bx = blockIdx.x * 32;           // source col block
    const int by = blockIdx.y * 32;           // source row block
    const int tx = threadIdx.x & 31;
    const int ty = threadIdx.x >> 5;          // 8 rows per pass
#pragma unroll
    for (int i = 0; i < 32; i += 8)
        tile[ty + i][tx] = sim[(size_t)(by + ty + i) * C + bx + tx];
    __syncthreads();
#pragma unroll
    for (int i = 0; i < 32; i += 8)
        simT[(size_t)(bx + ty + i) * C + by + tx] = (_Float16)tile[tx][ty + i];
}

// ---------------------------------------------------------------------------
// Kernel 1: posSim[B,C] = targets (0/1, f16) @ sim (f16 via simT), f32 out.
// 128 threads = 4 wave32 waves; each wave owns a 64x64 sub-tile
// = 4x4 fragments of v_wmma_f32_16x16x32_f16 (16 WMMA / K-step / wave).
// Double-buffered LDS staging.
// ---------------------------------------------------------------------------
__global__ __launch_bounds__(128) void possim_wmma_gemm(
    const float* __restrict__ tgt, const _Float16* __restrict__ simT,
    float* __restrict__ out, int C)
{
    // A: row-major rows x K (K contiguous). B: per-column K-contiguous.
    __shared__ _Float16 As [2][TM][TKS + LPAD];
    __shared__ _Float16 BsT[2][TN][TKS + LPAD];

    const int tid  = threadIdx.x;             // 0..127
    const int lane = tid & 31;
    const int wv   = tid >> 5;                // 0..3
    const int wm   = (wv & 1) * 64;
    const int wn   = (wv >> 1) * 64;
    const int hs   = lane >> 4;               // half-wave select
    const int l15  = lane & 15;

    const size_t mBase = (size_t)blockIdx.y * TM;
    const size_t nBase = (size_t)blockIdx.x * TN;

    v8f acc[4][4];
#pragma unroll
    for (int mi = 0; mi < 4; ++mi)
#pragma unroll
        for (int ni = 0; ni < 4; ++ni)
            acc[mi][ni] = (v8f){};

    // staging pointers: thread t owns one A row and one B (simT) row
    const float*    aSrc = tgt  + (mBase + tid) * (size_t)C;
    const _Float16* bSrc = simT + (nBase + tid) * (size_t)C;

    // ---- stage one K tile into LDS buffer `buf`
    auto stage = [&](int buf, int k0) {
        // A: 32 f32 -> 32 f16, four 16-B LDS stores
        const float4* g = reinterpret_cast<const float4*>(aSrc + k0);
#pragma unroll
        for (int q = 0; q < 4; ++q) {
            const float4 a = g[2 * q + 0];
            const float4 b = g[2 * q + 1];
            v8h h;
            h[0] = (_Float16)a.x; h[1] = (_Float16)a.y;
            h[2] = (_Float16)a.z; h[3] = (_Float16)a.w;
            h[4] = (_Float16)b.x; h[5] = (_Float16)b.y;
            h[6] = (_Float16)b.z; h[7] = (_Float16)b.w;
            *reinterpret_cast<v8h*>(&As[buf][tid][q * 8]) = h;
        }
        // B: 32 f16 straight copy (already transposed + converted)
        const uint4* gb = reinterpret_cast<const uint4*>(bSrc + k0);
        uint4*       db = reinterpret_cast<uint4*>(&BsT[buf][tid][0]);
#pragma unroll
        for (int q = 0; q < 4; ++q) db[q] = gb[q];
        if (k0 + 2 * TKS < C) {   // gfx1250 global_prefetch_b8, K+2 tile
            __builtin_prefetch(aSrc + k0 + 2 * TKS, 0, 1);
            __builtin_prefetch(bSrc + k0 + 2 * TKS, 0, 1);
        }
    };

    stage(0, 0);
    __syncthreads();

    for (int k0 = 0; k0 < C; k0 += TKS) {
        const int buf = (k0 / TKS) & 1;
        if (k0 + TKS < C) stage(buf ^ 1, k0 + TKS);

        // --- A fragments (ISA 7.12.2): lanes 0-15 K={0..7,16..23},
        //     lanes 16-31 K={8..15,24..31} -> two contiguous 8-half chunks
        v16h af[4];
#pragma unroll
        for (int mi = 0; mi < 4; ++mi) {
            const int r = wm + mi * 16 + l15;
            const v8h lo = *reinterpret_cast<const v8h*>(&As[buf][r][hs * 8]);
            const v8h hi = *reinterpret_cast<const v8h*>(&As[buf][r][16 + hs * 8]);
            af[mi] = __builtin_shufflevector(lo, hi, 0, 1, 2, 3, 4, 5, 6, 7,
                                             8, 9, 10, 11, 12, 13, 14, 15);
        }
        // --- B fragments: lanes 0-15 K={0..7,16..23}, lanes 16-31 K={8..15,24..31},
        //     K-contiguous in BsT -> two b128 LDS loads
        v16h bf[4];
#pragma unroll
        for (int ni = 0; ni < 4; ++ni) {
            const int cc = wn + ni * 16 + l15;
            const v8h lo = *reinterpret_cast<const v8h*>(&BsT[buf][cc][hs * 8]);
            const v8h hi = *reinterpret_cast<const v8h*>(&BsT[buf][cc][16 + hs * 8]);
            bf[ni] = __builtin_shufflevector(lo, hi, 0, 1, 2, 3, 4, 5, 6, 7,
                                             8, 9, 10, 11, 12, 13, 14, 15);
        }

#pragma unroll
        for (int mi = 0; mi < 4; ++mi)
#pragma unroll
            for (int ni = 0; ni < 4; ++ni)
                acc[mi][ni] = __builtin_amdgcn_wmma_f32_16x16x32_f16(
                    false, af[mi], false, bf[ni], (short)0, acc[mi][ni],
                    false, false);
        __syncthreads();
    }

    // --- C/D layout: VGPR r -> M = r + 8*hs, N = lane&15
#pragma unroll
    for (int mi = 0; mi < 4; ++mi) {
        const size_t row0 = mBase + wm + mi * 16 + hs * 8;
#pragma unroll
        for (int ni = 0; ni < 4; ++ni) {
            const size_t col = nBase + wn + ni * 16 + l15;
#pragma unroll
            for (int r = 0; r < 8; ++r)
                out[(row0 + r) * (size_t)C + col] = acc[mi][ni][r];
        }
    }
}

// ---------------------------------------------------------------------------
// Kernel 2: per-row Gumbel-top-k negative sampling + masked BCE.
// One 256-thread block per row; each thread owns C/256 (=8) columns.
// ---------------------------------------------------------------------------
__device__ __forceinline__ unsigned long long splitmix64(unsigned long long x) {
    x += 0x9E3779B97F4A7C15ull;
    x = (x ^ (x >> 30)) * 0xBF58476D1CE4E5B9ull;
    x = (x ^ (x >> 27)) * 0x94D049BB133111EBull;
    return x ^ (x >> 31);
}

__global__ __launch_bounds__(256) void rowselect_kernel(
    const float* __restrict__ logits, const float* __restrict__ tgt,
    const float* __restrict__ posSim, float* __restrict__ sums, int C)
{
    const int row = blockIdx.x;
    const int tid = threadIdx.x;
    int per = C >> 8;
    if (per > 8) per = 8;                  // register arrays sized for C<=2048
    const size_t base = (size_t)row * C;

    __shared__ float red[256];
    __shared__ int   icnt;

    // ---- num_pos (row sum of 0/1 targets)
    float np_l = 0.f;
    for (int i = 0; i < per; ++i) np_l += tgt[base + tid + (i << 8)];
    red[tid] = np_l; __syncthreads();
    for (int s = 128; s > 0; s >>= 1) {
        if (tid < s) red[tid] += red[tid + s];
        __syncthreads();
    }
    const float np = red[0];
    __syncthreads();

    // ---- sampling weights + Gumbel scores -> order-preserving uint keys
    unsigned key[8];
    unsigned finmask = 0;
    int nfin_l = 0;
    const float invnp = 1.f / fmaxf(np, 1.f);
    for (int i = 0; i < per; ++i) {
        const int c = tid + (i << 8);
        const float t  = tgt[base + c];
        const float ps = posSim[base + c];
        const float w  = (1.f - ps * invnp) * ((t == 0.f) ? 1.f : 0.f);
        unsigned k32 = 0u;
        if (w > 0.f) {
            unsigned long long h = splitmix64(
                ((unsigned long long)(row + 1) << 32) ^
                (unsigned long long)((unsigned)c * 2654435761u) ^ 0x2Aull);
            const float u = ((float)((h >> 40) & 0xFFFFFFu) + 0.5f) *
                            (1.f / 16777216.f);
            const float g = -logf(-logf(u));                 // Gumbel(0,1)
            const float s = logf(fmaxf(w, 1e-30f)) + g;
            const unsigned b = __float_as_uint(s);
            k32 = (b & 0x80000000u) ? ~b : (b | 0x80000000u);
            if (k32 == 0u) k32 = 1u;
            finmask |= (1u << i);
            ++nfin_l;
        }
        key[i] = k32;
    }
    red[tid] = (float)nfin_l; __syncthreads();
    for (int s = 128; s > 0; s >>= 1) {
        if (tid < s) red[tid] += red[tid + s];
        __syncthreads();
    }
    const int nfin = (int)red[0];
    __syncthreads();

    int keff = (int)floorf(fmaxf(np, 1.f) * 5.f);   // NEG_SAMPLE_RATIO = 5
    if (keff > nfin) keff = nfin;

    // ---- bitwise k-th-largest: max T with count(key >= T) >= keff
    unsigned prefix = 0u;
    for (int bit = 31; bit >= 0; --bit) {
        const unsigned cand = prefix | (1u << bit);
        int loc = 0;
        for (int i = 0; i < per; ++i)
            if ((finmask >> i) & 1u) loc += (key[i] >= cand) ? 1 : 0;
        if (tid == 0) icnt = 0;
        __syncthreads();
        if (loc) atomicAdd(&icnt, loc);
        __syncthreads();
        if (keff > 0 && icnt >= keff) prefix = cand;
        __syncthreads();
    }

    // ---- masked stable BCE-with-logits, accumulate (sum, count)
    float lsum = 0.f, lcnt = 0.f;
    for (int i = 0; i < per; ++i) {
        const int c = tid + (i << 8);
        const float t  = tgt[base + c];
        const float lg = logits[base + c];
        const bool  fin = (finmask >> i) & 1u;
        const bool  sel = (keff > 0) && fin && (key[i] >= prefix);
        const float mask = t + (sel ? 1.f : 0.f);
        const float bce  = fmaxf(lg, 0.f) - lg * t + log1pf(expf(-fabsf(lg)));
        lsum += bce * mask;
        lcnt += mask;
    }
    red[tid] = lsum; __syncthreads();
    for (int s = 128; s > 0; s >>= 1) {
        if (tid < s) red[tid] += red[tid + s];
        __syncthreads();
    }
    const float bsum = red[0];
    __syncthreads();
    red[tid] = lcnt; __syncthreads();
    for (int s = 128; s > 0; s >>= 1) {
        if (tid < s) red[tid] += red[tid + s];
        __syncthreads();
    }
    if (tid == 0) {
        atomicAdd(&sums[0], bsum);
        atomicAdd(&sums[1], red[0]);
    }
}

__global__ void init_sums(float* sums) { sums[0] = 0.f; sums[1] = 0.f; }

__global__ void finalize_kernel(const float* sums, float* out) {
    out[0] = sums[0] / sums[1];
}

// ---------------------------------------------------------------------------
extern "C" void kernel_launch(void* const* d_in, const int* in_sizes, int n_in,
                              void* d_out, int out_size, void* d_ws, size_t ws_size,
                              hipStream_t stream)
{
    const float* logits = (const float*)d_in[0];   // [B,C]
    const float* tgt    = (const float*)d_in[1];   // [B,C] 0/1
    const float* sim    = (const float*)d_in[2];   // [C,C]

    const int C = (int)(sqrt((double)in_sizes[2]) + 0.5);
    const int B = in_sizes[0] / C;

    float*     sums   = (float*)d_ws;                               // 2 f32
    float*     posSim = (float*)((char*)d_ws + 256);                // [B,C] f32
    _Float16*  simT   = (_Float16*)((char*)d_ws + 256 +
                                    (size_t)B * C * sizeof(float)); // [C,C] f16

    init_sums<<<1, 1, 0, stream>>>(sums);

    dim3 tg(C / 32, C / 32);
    transpose_f16_kernel<<<tg, 256, 0, stream>>>(sim, simT, C);

    dim3 gg(C / TN, B / TM);
    possim_wmma_gemm<<<gg, 128, 0, stream>>>(tgt, simT, posSim, C);

    rowselect_kernel<<<B, 256, 0, stream>>>(logits, tgt, posSim, sums, C);

    finalize_kernel<<<1, 1, 0, stream>>>(sums, (float*)d_out);
}
